// BahdanauAttention_30940944400617
// MI455X (gfx1250) — matrix operand order repaired and checked by reading, not compiled, not run on hardware
//
#include <hip/hip_runtime.h>
#include <hip/hip_bf16.h>

#define B_   16
#define LQ_  1024
#define LK_  1024
#define DK_  256
#define DV_  256
#define WAVES 4
#define NKB  (LK_ / 32)          // 32 k-steps
#define SLABH (32 * DV_)         // halves per B-slab (16KB)

typedef __attribute__((ext_vector_type(16))) _Float16 v16h;
typedef __attribute__((ext_vector_type(8)))  _Float16 v8h;
typedef __attribute__((ext_vector_type(8)))  float    v8f;
typedef __attribute__((ext_vector_type(4)))  int      v4i;

// CDNA5 async copy: global -> LDS, 16B, tracked by ASYNCcnt (no VGPR data).
__device__ __forceinline__ void async_b128(unsigned lds_off,
                                           unsigned long long ga) {
  asm volatile("global_load_async_to_lds_b128 %0, %1, off"
               :: "v"(lds_off), "v"(ga) : "memory");
}
__device__ __forceinline__ void wait_async0() {
  asm volatile("s_wait_asynccnt 0x0" ::: "memory");
}

// ---------------------------------------------------------------------------
// Kernel 1: k_s[b,k] = dot(key[b,k,:], w)   (q_s cancels in softmax, skip it)
// ---------------------------------------------------------------------------
__global__ void ks_kernel(const float* __restrict__ key,
                          const float* __restrict__ w,
                          float* __restrict__ ks) {
  int idx = blockIdx.x * blockDim.x + threadIdx.x;   // b*LK + k
  if (idx >= B_ * LK_) return;
  const float4* kp = (const float4*)(key + (size_t)idx * DK_);
  const float4* wp = (const float4*)w;
  float acc = 0.0f;
#pragma unroll 8
  for (int i = 0; i < DK_ / 4; ++i) {
    float4 a = kp[i];
    float4 b = wp[i];
    acc += a.x * b.x + a.y * b.y + a.z * b.z + a.w * b.w;
  }
  ks[idx] = acc;
}

// ---------------------------------------------------------------------------
// Kernel 2: per-batch max over k, then e16[b,k] = (f16) exp(ks - max)
// ---------------------------------------------------------------------------
__global__ void exp_kernel(const float* __restrict__ ks,
                           _Float16* __restrict__ e16) {
  __shared__ float red[32];
  const int b = blockIdx.x;
  const int t = threadIdx.x;
  float v = ks[(size_t)b * LK_ + t];
  float m = v;
#pragma unroll
  for (int off = 16; off; off >>= 1) m = fmaxf(m, __shfl_xor(m, off));
  if ((t & 31) == 0) red[t >> 5] = m;
  __syncthreads();
  if (t < 32) {
    float x = red[t];
#pragma unroll
    for (int off = 16; off; off >>= 1) x = fmaxf(x, __shfl_xor(x, off));
    red[t] = x;
  }
  __syncthreads();
  float mb = red[0];
  e16[(size_t)b * LK_ + t] = (_Float16)__expf(v - mb);
}

// ---------------------------------------------------------------------------
// Kernel 3: V (f32) -> f16 pre-swizzled WMMA B-fragments (contiguous K order:
// lanes 0-15 K=0..15, lanes 16-31 K=16..31, N=lane&15, 16 halves per lane).
// ---------------------------------------------------------------------------
__global__ void vswizzle_kernel(const float* __restrict__ V,
                                _Float16* __restrict__ vb) {
  int t = blockIdx.x * blockDim.x + threadIdx.x;     // 2^18 threads total
  int lane = t & 31;
  int n    = (t >> 5) & 15;
  int kblk = (t >> 9) & (NKB - 1);
  int b    = t >> 14;
  int col  = n * 16 + (lane & 15);
  int k0   = kblk * 32 + ((lane & 16) ? 16 : 0);
  const float* src = V + ((size_t)b * LK_ + k0) * DV_ + col;
  v16h o;
#pragma unroll
  for (int e = 0; e < 16; ++e) o[e] = (_Float16)src[(size_t)e * DV_];
  *(v16h*)(vb + (size_t)t * 16) = o;
}

// ---------------------------------------------------------------------------
// Kernel 4: out[b,q,:] = (A @ V) / denom, A[q,k] = mask * e16[k].
// 4 waves/block; B-slabs double-buffered in LDS via async-to-LDS; fragment
// reads group-of-4 double-buffered in VGPRs so ds loads stay in flight
// across WMMAs (partial dscnt waits instead of full drains).
// ---------------------------------------------------------------------------
__global__ __launch_bounds__(WAVES * 32, 1) void
pav_kernel(const int* __restrict__ mask,
           const _Float16* __restrict__ e16,
           const _Float16* __restrict__ vb,
           float* __restrict__ out) {
  __shared__ _Float16 lse[LK_];                               // 2KB
  __shared__ __align__(16) _Float16 vslab[2][SLABH];          // 2 x 16KB
  const int b    = blockIdx.y;
  const int wave = threadIdx.x >> 5;
  const int lane = threadIdx.x & 31;
  const int q0   = (blockIdx.x * WAVES + wave) * 16;

  const _Float16* vbsrc = vb + (size_t)b * NKB * SLABH;       // this batch

  // Prologue: async-stage e16[b,:] (2KB) and B-slab 0 (16KB) into LDS.
  {
    unsigned eoff = (unsigned)(uintptr_t)(&lse[0]) + (unsigned)threadIdx.x * 16u;
    unsigned long long ega =
        (unsigned long long)(uintptr_t)(e16 + (size_t)b * LK_) +
        (unsigned long long)threadIdx.x * 16ull;
    async_b128(eoff, ega);

    unsigned soff = (unsigned)(uintptr_t)(&vslab[0][0]) +
                    (unsigned)threadIdx.x * 128u;
    unsigned long long sga = (unsigned long long)(uintptr_t)vbsrc +
                             (unsigned long long)threadIdx.x * 128ull;
#pragma unroll
    for (int i = 0; i < 8; ++i) async_b128(soff + i * 16u, sga + i * 16ull);
    wait_async0();
  }
  __syncthreads();

  v8f c[16];
#pragma unroll
  for (int n = 0; n < 16; ++n)
#pragma unroll
    for (int j = 0; j < 8; ++j) c[n][j] = 0.0f;

  const int  row  = lane & 15;
  const int  koff = (lane & 16) ? 16 : 0;                     // A-frag K offset
  const int* mrow = mask + ((size_t)b * LQ_ + q0 + row) * LK_;

  float dpart = 0.0f;

  // Prologue: mask vectors for kblk = 0 (register-pipelined stream)
  v4i m0 = *(const v4i*)(mrow + koff);
  v4i m1 = *(const v4i*)(mrow + koff + 4);
  v4i m2 = *(const v4i*)(mrow + koff + 8);
  v4i m3 = *(const v4i*)(mrow + koff + 12);

  for (int kblk = 0; kblk < NKB; ++kblk) {
    const int k = kblk * 32;

    // Kick async prefetch of the NEXT B-slab into the other LDS buffer.
    if (kblk + 1 < NKB) {
      unsigned soff = (unsigned)(uintptr_t)(&vslab[(kblk + 1) & 1][0]) +
                      (unsigned)threadIdx.x * 128u;
      unsigned long long sga =
          (unsigned long long)(uintptr_t)(vbsrc + (size_t)(kblk + 1) * SLABH) +
          (unsigned long long)threadIdx.x * 128ull;
#pragma unroll
      for (int i = 0; i < 8; ++i) async_b128(soff + i * 16u, sga + i * 16ull);
    }

    // Build A fragment from mask + e16 (contiguous K order: element s = K koff+s, as in the B fragments).
    v8h e0 = *(const v8h*)(lse + k + koff);
    v8h e1 = *(const v8h*)(lse + k + koff + 8);
    v16h a;
#pragma unroll
    for (int j = 0; j < 4; ++j) {
      _Float16 x0 = m0[j] ? e0[j]     : (_Float16)0.0f;
      _Float16 x1 = m1[j] ? e0[j + 4] : (_Float16)0.0f;
      _Float16 x2 = m2[j] ? e1[j]     : (_Float16)0.0f;
      _Float16 x3 = m3[j] ? e1[j + 4] : (_Float16)0.0f;
      a[j]      = x0;
      a[j + 4]  = x1;
      a[j + 8]  = x2;
      a[j + 12] = x3;
      dpart += (float)x0 + (float)x1 + (float)x2 + (float)x3;
    }

    // Issue next k-step's mask loads: a full WMMA chain of latency cover.
    v4i n0 = m0, n1 = m1, n2 = m2, n3 = m3;
    if (kblk + 1 < NKB) {
      const int kn = k + 32;
      n0 = *(const v4i*)(mrow + kn + koff);
      n1 = *(const v4i*)(mrow + kn + koff + 4);
      n2 = *(const v4i*)(mrow + kn + koff + 8);
      n3 = *(const v4i*)(mrow + kn + koff + 12);
    }

    // 16 WMMAs on the current LDS slab, fragments group-of-4 double-buffered:
    // group g+1's 8 ds_load_b128 are issued BEFORE group g's WMMAs consume
    // their slots, forcing distinct VGPRs and partial dscnt waits.
    const v16h* fp = (const v16h*)(&vslab[kblk & 1][0]) + lane;
    v16h buf[8];
#pragma unroll
    for (int i = 0; i < 4; ++i) buf[i] = fp[(size_t)i * 32];
#pragma unroll
    for (int g = 0; g < 4; ++g) {
      if (g + 1 < 4) {
#pragma unroll
        for (int i = 0; i < 4; ++i)
          buf[((g + 1) & 1) * 4 + i] = fp[(size_t)((g + 1) * 4 + i) * 32];
      }
#pragma unroll
      for (int i = 0; i < 4; ++i) {
        const int n = g * 4 + i;
        c[n] = __builtin_amdgcn_wmma_f32_16x16x32_f16(
            false, a, false, buf[(g & 1) * 4 + i], (short)0, c[n], false, false);
      }
    }

    m0 = n0; m1 = n1; m2 = n2; m3 = n3;

    // Next slab must have landed; publish across waves.
    if (kblk + 1 < NKB) wait_async0();
    __syncthreads();
  }

  // denom[row] = halves from lanes L and L^16 (disjoint K partitions)
  float denom = dpart + __shfl_xor(dpart, 16);
  float inv   = 1.0f / denom;                         // lane L holds inv(row L&15)

  float* obase = out + ((size_t)b * LQ_ + q0) * DV_;
#pragma unroll
  for (int r = 0; r < 8; ++r) {
    int m = r + ((lane & 16) ? 8 : 0);                // C/D layout: VGPR r -> row
    float ir = __shfl(inv, m);
    float* op = obase + (size_t)m * DV_ + row;
#pragma unroll
    for (int n = 0; n < 16; ++n) op[n * 16] = c[n][r] * ir;
  }
}

// ---------------------------------------------------------------------------
extern "C" void kernel_launch(void* const* d_in, const int* in_sizes, int n_in,
                              void* d_out, int out_size, void* d_ws, size_t ws_size,
                              hipStream_t stream) {
  (void)in_sizes; (void)n_in; (void)out_size; (void)ws_size;
  // setup_inputs order: query, key, value, w, mask   (query cancels -> unused)
  const float* key  = (const float*)d_in[1];
  const float* val  = (const float*)d_in[2];
  const float* w    = (const float*)d_in[3];
  const int*   mask = (const int*)d_in[4];
  float* out = (float*)d_out;

  char* ws = (char*)d_ws;
  float*    ks  = (float*)ws;                             // 64 KB
  _Float16* e16 = (_Float16*)(ws + 64 * 1024);            // 32 KB
  _Float16* vb  = (_Float16*)(ws + 96 * 1024);            // 8 MB swizzled f16 V

  ks_kernel<<<dim3((B_ * LK_) / 256), 256, 0, stream>>>(key, w, ks);
  exp_kernel<<<dim3(B_), 1024, 0, stream>>>(ks, e16);
  vswizzle_kernel<<<dim3((B_ * NKB * 16 * 32) / 256), 256, 0, stream>>>(val, vb);
  pav_kernel<<<dim3(LQ_ / (16 * WAVES), B_), WAVES * 32, 0, stream>>>(mask, e16, vb, out);
}
